// MatMulFreeLanguageModel_19662360281632
// MI455X (gfx1250) — compile-verified, matrix operand order checked by reading, not run on hardware
//
#include <hip/hip_runtime.h>
#include <hip/hip_bf16.h>

// ---------------------------------------------------------------------------
// Shapes (fixed by the reference)
// ---------------------------------------------------------------------------
#define BB   32
#define TT   256
#define EE   1024
#define VV   256
#define HH   16
#define HD   64
#define BT   (BB * TT)        // 8192 rows

typedef __attribute__((ext_vector_type(16))) _Float16 v16h;
typedef __attribute__((ext_vector_type(8)))  float    v8f;
typedef __attribute__((ext_vector_type(4)))  float    f32x4;
typedef __attribute__((ext_vector_type(2)))  unsigned u32x2;
typedef __attribute__((ext_vector_type(4)))  unsigned u32x4;
typedef __attribute__((ext_vector_type(4)))  int      i32x4;
typedef __attribute__((ext_vector_type(8)))  int      i32x8;

// ---------------------------------------------------------------------------
// Small device helpers
// ---------------------------------------------------------------------------
__device__ __forceinline__ float sigm(float x) { return 1.0f / (1.0f + __expf(-x)); }
__device__ __forceinline__ float silu(float x) { return x * sigm(x); }

__device__ __forceinline__ void atomicMaxF(float* p, float v) {
  // v >= 0 always here; non-negative float bit pattern preserves ordering as uint
  atomicMax(reinterpret_cast<unsigned*>(p), __float_as_uint(v));
}

// Load 16 halves matching CDNA5 16-bit WMMA operand layout:
// halves 0..7 from p[0..7], halves 8..15 from p[16..23]  (two b128 loads)
__device__ __forceinline__ v16h load_op16(const _Float16* p) {
  v16h r;
  reinterpret_cast<f32x4*>(&r)[0] = *reinterpret_cast<const f32x4*>(p);
  reinterpret_cast<f32x4*>(&r)[1] = *reinterpret_cast<const f32x4*>(p + 16);
  return r;
}

// ---------------------------------------------------------------------------
// Elementwise / prep kernels
// ---------------------------------------------------------------------------
__global__ void k_zero(float* p, size_t n) {
  size_t i = (size_t)blockIdx.x * blockDim.x + threadIdx.x;
  if (i < n) p[i] = 0.0f;
}

__global__ void k_copy_scale(float* dst, const float* src, int n, float s) {
  int i = blockIdx.x * blockDim.x + threadIdx.x;
  if (i < n) dst[i] = src[i] * s;
}

__global__ void k_cast16(_Float16* dst, const float* src, size_t n) {
  size_t i = (size_t)blockIdx.x * blockDim.x + threadIdx.x;
  if (i < n) dst[i] = (_Float16)src[i];
}

// x[b,t,e] = input_ids[b,t,e] * emb[e, t]   (emb: [E, V], V == T)
__global__ void k_scale_emb(float* x, const float* in, const float* emb) {
  size_t i = (size_t)blockIdx.x * blockDim.x + threadIdx.x;
  if (i >= (size_t)BT * EE) return;
  int e = (int)(i & (EE - 1));
  int t = (int)((i >> 10) & (TT - 1));
  x[i] = in[i] * emb[(size_t)e * VV + t];
}

// Transpose+convert weight: dst[k*ld + colOff + n] = maybe_sign(W[n*K + k])
__global__ void k_prep_w(_Float16* dst, const float* W, int N, int K, int ld,
                         int colOff, int doSign) {
  size_t i = (size_t)blockIdx.x * blockDim.x + threadIdx.x;
  if (i >= (size_t)N * K) return;
  int n = (int)(i / K);
  int k = (int)(i % K);
  float w = W[i];
  if (doSign) w = (w > 0.0f) ? 1.0f : ((w < 0.0f) ? -1.0f : 0.0f);
  dst[(size_t)k * ld + colOff + n] = (_Float16)w;
}

// ---------------------------------------------------------------------------
// Mean-centered layernorm per row (E=1024) + absmax (per-t or global slot)
// ---------------------------------------------------------------------------
__global__ void k_rmsn(const float* __restrict__ x, float* __restrict__ y,
                       float* amax, int perT) {
  __shared__ float red[256];
  const int row = blockIdx.x;
  const int tid = threadIdx.x;
  const float* xr = x + (size_t)row * EE;
  float* yr = y + (size_t)row * EE;

  float s = 0.0f;
  for (int i = tid; i < EE; i += 256) s += xr[i];
  red[tid] = s; __syncthreads();
  for (int o = 128; o; o >>= 1) { if (tid < o) red[tid] += red[tid + o]; __syncthreads(); }
  const float mu = red[0] * (1.0f / EE);
  __syncthreads();

  float v = 0.0f;
  for (int i = tid; i < EE; i += 256) { float d = xr[i] - mu; v += d * d; }
  red[tid] = v; __syncthreads();
  for (int o = 128; o; o >>= 1) { if (tid < o) red[tid] += red[tid + o]; __syncthreads(); }
  const float var = red[0] * (1.0f / EE);
  const float r = rsqrtf(fmaxf(var + 1e-8f, 1e-10f));
  __syncthreads();

  float m = 0.0f;
  for (int i = tid; i < EE; i += 256) {
    float o = r * (xr[i] - mu);
    yr[i] = o;
    m = fmaxf(m, fabsf(o));
  }
  red[tid] = m; __syncthreads();
  for (int o = 128; o; o >>= 1) { if (tid < o) red[tid] = fmaxf(red[tid], red[tid + o]); __syncthreads(); }
  if (tid == 0) atomicMaxF(&amax[perT ? (row & (TT - 1)) : 0], red[0]);
}

// ---------------------------------------------------------------------------
// Fake-quant: q = clip(round(x/s), -128, 127) * s,  s = amax/127 (or 1 if 0)
// ---------------------------------------------------------------------------
__global__ void k_quant(const float* __restrict__ x, const float* amax, int perT,
                        _Float16* o16, float* o32, float* nxt, size_t n) {
  __shared__ float red[256];
  size_t i = (size_t)blockIdx.x * 256 + threadIdx.x;
  float q = 0.0f;
  if (i < n) {
    float m = amax[perT ? (int)((i >> 10) & (TT - 1)) : 0];
    float s = (m == 0.0f) ? 1.0f : m * (1.0f / 127.0f);
    q = rintf(x[i] / s);
    q = fminf(fmaxf(q, -128.0f), 127.0f) * s;
    if (o16) o16[i] = (_Float16)q;
    if (o32) o32[i] = q;
  }
  if (nxt) {
    red[threadIdx.x] = fabsf(q); __syncthreads();
    for (int o = 128; o; o >>= 1) {
      if (threadIdx.x < o) red[threadIdx.x] = fmaxf(red[threadIdx.x], red[threadIdx.x + o]);
      __syncthreads();
    }
    if (threadIdx.x == 0) atomicMaxF(nxt, red[0]);
  }
}

// p = g * u from GU [BT, 2048], plus absmax of p
__global__ void k_glu_mul(float* __restrict__ p, const float* __restrict__ GU,
                          float* nxt) {
  __shared__ float red[256];
  size_t i = (size_t)blockIdx.x * 256 + threadIdx.x;
  float q = 0.0f;
  if (i < (size_t)BT * EE) {
    size_t row = i >> 10;
    int c = (int)(i & (EE - 1));
    q = GU[row * 2048 + c] * GU[row * 2048 + 1024 + c];
    p[i] = q;
  }
  red[threadIdx.x] = fabsf(q); __syncthreads();
  for (int o = 128; o; o >>= 1) {
    if (threadIdx.x < o) red[threadIdx.x] = fmaxf(red[threadIdx.x], red[threadIdx.x + o]);
    __syncthreads();
  }
  if (threadIdx.x == 0) atomicMaxF(nxt, red[0]);
}

// Elementwise scan over t: h = f*h + (1-f)*c ; out = h*g
__global__ void k_scan(float* __restrict__ xout, const float* __restrict__ FCG) {
  int i = blockIdx.x * blockDim.x + threadIdx.x;   // over B*E = 32768
  if (i >= BB * EE) return;
  int b = i >> 10;
  int e = i & (EE - 1);
  float h = 0.0f;
  for (int t = 0; t < TT; ++t) {
    size_t base = (size_t)(b * TT + t) * 3072;
    float f = FCG[base + e];
    float c = FCG[base + 1024 + e];
    float g = FCG[base + 2048 + e];
    h = f * h + (1.0f - f) * c;
    xout[(size_t)(b * TT + t) * EE + e] = h * g;
  }
}

// ---------------------------------------------------------------------------
// WMMA GEMM:  C[M,N] = act(A[M,K] @ B[K,N] + bias[N]) (+ resid[M,N])
// Block: 256 threads = 8 waves; tile BM=64 x BN=128, BK=32.
// Wave (wm = wave&3, nh = wave>>2) computes one 16-row A fragment reused
// across four 16-col B fragments -> 4 v_wmma per wave per k-step.
// ACT: 0=none 1=sigmoid 2=silu 3=mlgru split(sig|silu|sig) 4=glu split(silu|none)
// ---------------------------------------------------------------------------
template <int ACT, bool RESID>
__global__ __launch_bounds__(256) void k_gemm(
    const _Float16* __restrict__ A, const _Float16* __restrict__ Bw,
    const float* __restrict__ bias, const float* __restrict__ resid,
    float* __restrict__ C, int M, int N, int K, int actSplit) {
  __shared__ _Float16 As[64][32];    // [m][k]  4 KB
  __shared__ _Float16 Bs[128][32];   // [n][k]  8 KB (transposed stage)

  const int tid  = threadIdx.x;
  const int wave = tid >> 5;
  const int lane = tid & 31;
  const int wm   = wave & 3;             // m-tile 0..3
  const int nh   = wave >> 2;            // n half 0..1
  const int m0   = blockIdx.y * 64;
  const int n0   = blockIdx.x * 128;

  const int mrow = lane & 15;
  const int kb   = (lane < 16) ? 0 : 8;

  v8f acc[4] = {};

  for (int k0 = 0; k0 < K; k0 += 32) {
    // Stage A: 64x32 halves, one b128 per thread
    {
      int r = tid >> 2;
      int c = (tid & 3) * 8;
      const _Float16* src = A + (size_t)(m0 + r) * K + k0 + c;
      *reinterpret_cast<f32x4*>(&As[r][c]) = *reinterpret_cast<const f32x4*>(src);
      if (k0 + 32 < K) __builtin_prefetch(src + 32, 0, 0);
    }
    // Stage B transposed: read Bw[k][n] coalesced over n, write Bs[n][k]
    {
      int k = tid >> 3;
      int nn = (tid & 7) * 16;
      const _Float16* src = Bw + (size_t)(k0 + k) * N + n0 + nn;
#pragma unroll
      for (int j = 0; j < 16; ++j) Bs[nn + j][k] = src[j];
      if (k0 + 32 < K) __builtin_prefetch(src + (size_t)32 * N, 0, 0);
    }
    __syncthreads();

    v16h a = load_op16(&As[wm * 16 + mrow][kb]);
#pragma unroll
    for (int ti = 0; ti < 4; ++ti) {
      v16h b = load_op16(&Bs[nh * 64 + ti * 16 + mrow][kb]);
      acc[ti] = __builtin_amdgcn_wmma_f32_16x16x32_f16(false, a, false, b,
                                                       (short)0, acc[ti], false, false);
    }
    __syncthreads();
  }

  // Epilogue: C layout — lane n = lane&15; VGPR r -> m = (lane<16 ? 0 : 8) + r
  const int mb = m0 + wm * 16 + ((lane < 16) ? 0 : 8);
#pragma unroll
  for (int ti = 0; ti < 4; ++ti) {
    const int cn = n0 + nh * 64 + ti * 16 + (lane & 15);
    const float bv = bias ? bias[cn] : 0.0f;
#pragma unroll
    for (int r = 0; r < 8; ++r) {
      const int cm = mb + r;
      float v = acc[ti][r] + bv;
      if (ACT == 1) v = sigm(v);
      else if (ACT == 2) v = silu(v);
      else if (ACT == 3) { int seg = cn / actSplit; v = (seg == 1) ? silu(v) : sigm(v); }
      else if (ACT == 4) { if (cn < actSplit) v = silu(v); }
      if (RESID) v += resid[(size_t)cm * N + cn];
      C[(size_t)cm * N + cn] = v;
    }
  }
}

// ---------------------------------------------------------------------------
// Fused attention per (b, h): scores = QK^T/8 + 1 -> softmax -> @V
// K slab staged to LDS via the Tensor Data Mover (TDM) when available;
// V^T staged manually (TDM has no transpose). Head-averaged weights are
// accumulated into attnw [B, T, T]. Dynamic LDS (~91 KB).
// ---------------------------------------------------------------------------
__global__ __launch_bounds__(256) void k_attn(
    const _Float16* __restrict__ qh, const _Float16* __restrict__ kh,
    const _Float16* __restrict__ vh, float* __restrict__ aout,
    float* __restrict__ attnw) {
  extern __shared__ char smem[];
  _Float16* Ks = reinterpret_cast<_Float16*>(smem);        // [256][64]
  _Float16* Vt = Ks + TT * HD;                             // [64][256] (transposed)
  float*    sc = reinterpret_cast<float*>(Vt + HD * TT);   // [16][256]
  _Float16* wh = reinterpret_cast<_Float16*>(sc + 16 * TT);// [16][256]
  float*   red = reinterpret_cast<float*>(wh + 16 * TT);   // [16][16]

  const int b = blockIdx.x / HH;
  const int h = blockIdx.x % HH;
  const int tid = threadIdx.x, lane = tid & 31, wave = tid >> 5;
  const int mrow = lane & 15;
  const int kb = (lane < 16) ? 0 : 8;

  const _Float16* Q = qh + (size_t)b * TT * EE + h * HD;

#if __has_builtin(__builtin_amdgcn_tensor_load_to_lds)
  // --- TDM: DMA the 256x64 f16 K tile (rows of a [8192,1024] tensor) into Ks
  if (wave == 0) {
    unsigned long long ga =
        (unsigned long long)(uintptr_t)(kh + (size_t)b * TT * EE + (size_t)h * HD);
    unsigned lds = (unsigned)(uintptr_t)(void*)Ks;   // aperture: low 32 bits = LDS offset
    u32x4 g0 = {};
    g0[0] = 1u;                                      // count=1, user mode
    g0[1] = lds;                                     // lds_addr
    g0[2] = (unsigned)(ga & 0xFFFFFFFFu);            // global_addr[31:0]
    g0[3] = (unsigned)((ga >> 32) & 0x01FFFFFFu) | (2u << 30);  // addr[56:32] | type=2
    i32x8 g1 = {};
    g1[0] = (1 << 16);                               // data_size=1 (2 bytes)
    g1[1] = (int)((1024u & 0xFFFFu) << 16);          // tensor_dim0 = 1024 (bits 79:48)
    g1[2] = (int)(((1024u >> 16) & 0xFFFFu) | ((8192u & 0xFFFFu) << 16)); // dim0 hi | dim1 lo
    g1[3] = (int)(((8192u >> 16) & 0xFFFFu) | (64u << 16)); // dim1 hi | tile_dim0=64
    g1[4] = (int)256u;                               // tile_dim1=256, tile_dim2=0
    g1[5] = 1024;                                    // tensor_dim0_stride (elems)
    g1[6] = 0;
    g1[7] = 0;
    i32x4 gz4 = {};
#if __clang_major__ >= 23
    i32x8 gz8 = {};
    __builtin_amdgcn_tensor_load_to_lds(g0, g1, gz4, gz4, gz8, 0);
#else
    __builtin_amdgcn_tensor_load_to_lds(g0, g1, gz4, gz4, 0);
#endif
  }
  // --- meanwhile stage V transposed by all waves
  for (int i = tid; i < TT * HD; i += 256) {
    int key = i >> 6, d = i & (HD - 1);
    Vt[d * TT + key] = vh[((size_t)b * TT + key) * EE + h * HD + d];
  }
  if (wave == 0) __builtin_amdgcn_s_wait_tensorcnt(0);
  __syncthreads();
#else
  for (int i = tid; i < TT * HD; i += 256) {
    int key = i >> 6, d = i & (HD - 1);
    size_t g = ((size_t)b * TT + key) * EE + h * HD + d;
    Ks[key * HD + d] = kh[g];
    Vt[d * TT + key] = vh[g];
  }
  __syncthreads();
#endif

  for (int qt = 0; qt < TT / 16; ++qt) {
    const int q0 = qt * 16;
    // --- scores: two WMMA k-steps (d 0..31, 32..63), 8 waves x 2 col tiles
    const _Float16* pq = Q + (size_t)(q0 + mrow) * EE;
    v16h aq0 = load_op16(pq + kb);
    v16h aq1 = load_op16(pq + 32 + kb);
#pragma unroll
    for (int u = 0; u < 2; ++u) {
      int n0 = (wave * 2 + u) * 16;
      v8f c = {};
      v16h b0 = load_op16(&Ks[(n0 + mrow) * HD + kb]);
      c = __builtin_amdgcn_wmma_f32_16x16x32_f16(false, aq0, false, b0, (short)0, c, false, false);
      v16h b1 = load_op16(&Ks[(n0 + mrow) * HD + 32 + kb]);
      c = __builtin_amdgcn_wmma_f32_16x16x32_f16(false, aq1, false, b1, (short)0, c, false, false);
      int nn = n0 + (lane & 15);
      int mb = (lane < 16) ? 0 : 8;
#pragma unroll
      for (int r = 0; r < 8; ++r)
        sc[(mb + r) * TT + nn] = c[r] * 0.125f + 1.0f;   // scale + all-ones mask
    }
    __syncthreads();

    // --- softmax over each of 16 rows (16 threads per row, 16 cols each)
    const int row = tid >> 4, sub = tid & 15;
    float* srow = sc + row * TT;
    float pm = -1e30f;
#pragma unroll
    for (int j = 0; j < 16; ++j) pm = fmaxf(pm, srow[sub * 16 + j]);
    red[row * 16 + sub] = pm; __syncthreads();
    float rmax = red[row * 16];
#pragma unroll
    for (int j = 1; j < 16; ++j) rmax = fmaxf(rmax, red[row * 16 + j]);
    __syncthreads();
    float ps = 0.0f;
#pragma unroll
    for (int j = 0; j < 16; ++j) {
      float e = __expf(srow[sub * 16 + j] - rmax);
      srow[sub * 16 + j] = e;
      ps += e;
    }
    red[row * 16 + sub] = ps; __syncthreads();
    float rsum = 0.0f;
#pragma unroll
    for (int j = 0; j < 16; ++j) rsum += red[row * 16 + j];
    const float inv = 1.0f / rsum;
    __syncthreads();
#pragma unroll
    for (int j = 0; j < 16; ++j) {
      int cc = sub * 16 + j;
      float w = srow[cc] * inv;
      wh[row * TT + cc] = (_Float16)w;
      atomicAdd(attnw + ((size_t)b * TT + (q0 + row)) * TT + cc, w * (1.0f / HH));
    }
    __syncthreads();

    // --- out tile = wh[16,256] @ V[256,64]; waves 0..3 own 16-wide d tiles
    if (wave < 4) {
      v8f c = {};
      for (int ks = 0; ks < TT; ks += 32) {
        v16h aw = load_op16(&wh[mrow * TT + ks + kb]);
        v16h bvv = load_op16(&Vt[(wave * 16 + mrow) * TT + ks + kb]);
        c = __builtin_amdgcn_wmma_f32_16x16x32_f16(false, aw, false, bvv, (short)0, c, false, false);
      }
      int dd = wave * 16 + (lane & 15);
      int mb = (lane < 16) ? 0 : 8;
#pragma unroll
      for (int r = 0; r < 8; ++r)
        aout[((size_t)b * TT + q0 + mb + r) * EE + h * HD + dd] = c[r];
    }
    __syncthreads();
  }
}

// ---------------------------------------------------------------------------
// Host-side orchestration
// ---------------------------------------------------------------------------
static inline dim3 grid1(size_t n) { return dim3((unsigned)((n + 255) / 256)); }

extern "C" void kernel_launch(void* const* d_in, const int* in_sizes, int n_in,
                              void* d_out, int out_size, void* d_ws, size_t ws_size,
                              hipStream_t stream) {
  (void)in_sizes; (void)n_in; (void)out_size; (void)ws_size;
  const float* input_ids = (const float*)d_in[0];
  const float* prev      = (const float*)d_in[1];
  const float* emb       = (const float*)d_in[2];
  const float* W_f = (const float*)d_in[3];
  const float* W_c = (const float*)d_in[4];
  const float* W_g = (const float*)d_in[5];
  const float* b_f = (const float*)d_in[6];
  const float* b_c = (const float*)d_in[7];
  const float* b_g = (const float*)d_in[8];
  const float* Wg2 = (const float*)d_in[9];
  const float* Wu  = (const float*)d_in[10];
  const float* Wd  = (const float*)d_in[11];
  const float* bg2 = (const float*)d_in[12];
  const float* bu  = (const float*)d_in[13];
  const float* bd  = (const float*)d_in[14];
  const float* in_proj_w = (const float*)d_in[15];
  const float* in_proj_b = (const float*)d_in[16];
  const float* out_proj_w = (const float*)d_in[17];
  const float* out_proj_b = (const float*)d_in[18];
  const float* W_out = (const float*)d_in[19];
  const float* b_out = (const float*)d_in[20];

  float* fOut  = (float*)d_out;           // final: [BT, V]
  float* attnw = fOut + (size_t)BT * VV;  // attn weights: [B, T, T]

  // ---- workspace layout ----
  char* w = (char*)d_ws;
  size_t off = 0;
  auto take = [&](size_t bytes) { char* p = w + off; off += (bytes + 255) & ~(size_t)255; return p; };
  float* scal  = (float*)take(1024 * 4);                  // slots[8] + tmax[256]
  float* tmax  = scal + 8;
  float* xbuf  = (float*)take((size_t)BT * EE * 4);
  float* ybuf  = (float*)take((size_t)BT * EE * 4);
  float* qbuf  = (float*)take((size_t)BT * EE * 4);
  float* big   = (float*)take((size_t)BT * 3072 * 4);     // FCG / GU / q,k,v f32
  _Float16* a16    = (_Float16*)take((size_t)BT * EE * 2);
  _Float16* prev16 = (_Float16*)take((size_t)BT * EE * 2);
  _Float16* qh     = (_Float16*)take((size_t)BT * EE * 2);
  _Float16* kh     = (_Float16*)take((size_t)BT * EE * 2);
  _Float16* vh     = (_Float16*)take((size_t)BT * EE * 2);
  _Float16* w3     = (_Float16*)take((size_t)EE * 3072 * 2);
  _Float16* wglu   = (_Float16*)take((size_t)EE * 2048 * 2);
  _Float16* wd16   = (_Float16*)take((size_t)EE * EE * 2);
  _Float16* wq16   = (_Float16*)take((size_t)EE * EE * 2);
  _Float16* wk16   = (_Float16*)take((size_t)EE * EE * 2);
  _Float16* wv16   = (_Float16*)take((size_t)EE * EE * 2);
  _Float16* wop16  = (_Float16*)take((size_t)EE * EE * 2);
  _Float16* wout16 = (_Float16*)take((size_t)EE * VV * 2);
  float* bias3   = (float*)take(3072 * 4);
  float* biasglu = (float*)take(2048 * 4);
  float* bout2   = (float*)take(VV * 4);

  const size_t NE  = (size_t)BT * EE;
  const size_t NEK = (size_t)EE * EE;

  // ---- Stage 0: zero reductions + attnw output region; prep weights/biases
  k_zero<<<grid1(264), 256, 0, stream>>>(scal, 264);
  k_zero<<<grid1((size_t)BB * TT * TT), 256, 0, stream>>>(attnw, (size_t)BB * TT * TT);
  k_prep_w<<<grid1(NEK), 256, 0, stream>>>(w3, W_f, EE, EE, 3072, 0, 1);
  k_prep_w<<<grid1(NEK), 256, 0, stream>>>(w3, W_c, EE, EE, 3072, 1024, 1);
  k_prep_w<<<grid1(NEK), 256, 0, stream>>>(w3, W_g, EE, EE, 3072, 2048, 1);
  k_prep_w<<<grid1(NEK), 256, 0, stream>>>(wglu, Wg2, EE, EE, 2048, 0, 1);
  k_prep_w<<<grid1(NEK), 256, 0, stream>>>(wglu, Wu, EE, EE, 2048, 1024, 1);
  k_prep_w<<<grid1(NEK), 256, 0, stream>>>(wd16, Wd, EE, EE, 1024, 0, 1);
  k_prep_w<<<grid1(NEK), 256, 0, stream>>>(wq16, in_proj_w, EE, EE, 1024, 0, 0);
  k_prep_w<<<grid1(NEK), 256, 0, stream>>>(wk16, in_proj_w + NEK, EE, EE, 1024, 0, 0);
  k_prep_w<<<grid1(NEK), 256, 0, stream>>>(wv16, in_proj_w + 2 * NEK, EE, EE, 1024, 0, 0);
  k_prep_w<<<grid1(NEK), 256, 0, stream>>>(wop16, out_proj_w, EE, EE, 1024, 0, 0);
  k_prep_w<<<grid1((size_t)VV * EE), 256, 0, stream>>>(wout16, W_out, VV, EE, 256, 0, 0);
  k_copy_scale<<<grid1(EE), 256, 0, stream>>>(bias3, b_f, EE, 1.0f);
  k_copy_scale<<<grid1(EE), 256, 0, stream>>>(bias3 + 1024, b_c, EE, 1.0f);
  k_copy_scale<<<grid1(EE), 256, 0, stream>>>(bias3 + 2048, b_g, EE, 1.0f);
  k_copy_scale<<<grid1(EE), 256, 0, stream>>>(biasglu, bg2, EE, 1.0f);
  k_copy_scale<<<grid1(EE), 256, 0, stream>>>(biasglu + 1024, bu, EE, 1.0f);
  k_copy_scale<<<grid1(VV), 256, 0, stream>>>(bout2, b_out, VV, 2.0f);

  // ---- Stage 1: x = input_ids * emb.T
  k_scale_emb<<<grid1(NE), 256, 0, stream>>>(xbuf, input_ids, emb);

  // ---- Stage 2-3: per-t layernorm + per-t quant -> a16
  k_rmsn<<<BT, 256, 0, stream>>>(xbuf, ybuf, tmax, 1);
  k_quant<<<grid1(NE), 256, 0, stream>>>(ybuf, tmax, 1, a16, nullptr, nullptr, NE);

  // ---- Stage 4: gates GEMM [8192,1024]x[1024,3072] (sig|silu|sig epilogue)
  k_gemm<3, false><<<dim3(3072 / 128, BT / 64), 256, 0, stream>>>(
      a16, w3, bias3, nullptr, big, BT, 3072, EE, 1024);

  // ---- Stage 5: elementwise recurrence scan -> xbuf
  k_scan<<<grid1((size_t)BB * EE), 256, 0, stream>>>(xbuf, big);

  // ---- Stage 6-8: GLU input: rmsn (global absmax slot0), quant, quant again
  k_rmsn<<<BT, 256, 0, stream>>>(xbuf, ybuf, &scal[0], 0);
  k_quant<<<grid1(NE), 256, 0, stream>>>(ybuf, &scal[0], 0, nullptr, qbuf, &scal[1], NE);
  k_quant<<<grid1(NE), 256, 0, stream>>>(qbuf, &scal[1], 0, a16, nullptr, nullptr, NE);

  // ---- Stage 9: GLU g|u GEMM (silu|none epilogue) -> big[:, :2048]
  k_gemm<4, false><<<dim3(2048 / 128, BT / 64), 256, 0, stream>>>(
      a16, wglu, biasglu, nullptr, big, BT, 2048, EE, 1024);

  // ---- Stage 10-12: p = g*u (+absmax slot2), quant, down GEMM -> xbuf
  k_glu_mul<<<grid1(NE), 256, 0, stream>>>(qbuf, big, &scal[2]);
  k_quant<<<grid1(NE), 256, 0, stream>>>(qbuf, &scal[2], 0, a16, nullptr, nullptr, NE);
  k_gemm<0, false><<<dim3(EE / 128, BT / 64), 256, 0, stream>>>(
      a16, wd16, bd, nullptr, xbuf, BT, EE, EE, 0);

  // ---- Stage 13: out = quant(rmsn(x)) -> a16  (query input)
  k_rmsn<<<BT, 256, 0, stream>>>(xbuf, ybuf, &scal[3], 0);
  k_quant<<<grid1(NE), 256, 0, stream>>>(ybuf, &scal[3], 0, a16, nullptr, nullptr, NE);

  // ---- Stage 14-15: Q/K/V projections
  k_gemm<0, false><<<dim3(EE / 128, BT / 64), 256, 0, stream>>>(
      a16, wq16, in_proj_b, nullptr, big, BT, EE, EE, 0);
  k_cast16<<<grid1(NE), 256, 0, stream>>>(qh, big, NE);
  k_cast16<<<grid1(NE), 256, 0, stream>>>(prev16, prev, NE);
  k_gemm<0, false><<<dim3(EE / 128, BT / 64), 256, 0, stream>>>(
      prev16, wk16, in_proj_b + 1024, nullptr, big + NE, BT, EE, EE, 0);
  k_cast16<<<grid1(NE), 256, 0, stream>>>(kh, big + NE, NE);
  k_gemm<0, false><<<dim3(EE / 128, BT / 64), 256, 0, stream>>>(
      prev16, wv16, in_proj_b + 2048, nullptr, big + 2 * NE, BT, EE, EE, 0);
  k_cast16<<<grid1(NE), 256, 0, stream>>>(vh, big + 2 * NE, NE);

  // ---- Stage 16: fused attention (writes aout=xbuf, accumulates attnw)
  const size_t attn_lds = (size_t)TT * HD * 2   // Ks
                        + (size_t)HD * TT * 2   // Vt
                        + (size_t)16 * TT * 4   // scores
                        + (size_t)16 * TT * 2   // wh
                        + 16 * 16 * 4;          // red
  k_attn<<<BB * HH, 256, attn_lds, stream>>>(qh, kh, vh, xbuf, attnw);

  // ---- Stage 17: out_proj + residual(prev) -> ybuf
  k_cast16<<<grid1(NE), 256, 0, stream>>>(a16, xbuf, NE);
  k_gemm<0, true><<<dim3(EE / 128, BT / 64), 256, 0, stream>>>(
      a16, wop16, out_proj_b, prev, ybuf, BT, EE, EE, 0);

  // ---- Stage 18: final = out @ W_out.T + 2*b_out -> d_out
  k_cast16<<<grid1(NE), 256, 0, stream>>>(a16, ybuf, NE);
  k_gemm<0, false><<<dim3(VV / 128, BT / 64), 256, 0, stream>>>(
      a16, wout16, bout2, nullptr, fOut, BT, VV, EE, 0);
}
